// CausalMultiHeadAttention_44495861186749
// MI455X (gfx1250) — matrix operand order repaired and checked by reading, not compiled, not run on hardware
//
#include <hip/hip_runtime.h>
#include <hip/hip_bf16.h>

// ---------------------------------------------------------------------------
// Causal MHA for MI455X (gfx1250), bf16 WMMA pipeline.
//   B=4, S=2048, D=1024, H=16, HD=64.  All GEMMs via v_wmma_f32_16x16x32_bf16.
// ---------------------------------------------------------------------------

typedef __attribute__((ext_vector_type(16))) __bf16 v16bf;
typedef __attribute__((ext_vector_type(8)))  __bf16 v8bf;
typedef __attribute__((ext_vector_type(8)))  float  v8f;

union V16 { v16bf v; v8bf h[2]; };

constexpr int BB = 4;
constexpr int SS = 2048;
constexpr int DD = 1024;
constexpr int HH = 16;
constexpr int HD = 64;
constexpr int MM = BB * SS;          // 8192 rows for projection GEMMs

__device__ __forceinline__ unsigned short f2bf(float f) {
  unsigned u = __float_as_uint(f);
  unsigned r = u + 0x7FFFu + ((u >> 16) & 1u);   // round-to-nearest-even
  return (unsigned short)(r >> 16);
}

// ---------------------------------------------------------------------------
// fp32 -> bf16 cast
// ---------------------------------------------------------------------------
__global__ void cast_f32_to_bf16(const float* __restrict__ src,
                                 unsigned short* __restrict__ dst, int n) {
  int i = blockIdx.x * blockDim.x + threadIdx.x;
  if (i < n) dst[i] = f2bf(src[i]);
}

// ---------------------------------------------------------------------------
// GEMM: Y[M,N] = X[M,K] @ W[N,K]^T   (bf16 A/B, f32 accumulate)
//   Wave computes a 64x64 tile: 4 (M) x 4 (N) WMMA accumulators ->
//   16 WMMA per 16 b128 loads per 32-wide K step (1:1 issue ratio).
//   B fragments are streamed one at a time to cap register pressure.
//   MODE 0: f32 row-major [M,N] out (final projection -> d_out)
//   MODE 1: bf16 out scattered to [B,H,S,HD]  (Q, K)
//   MODE 2: bf16 out scattered to [B,H,HD,S]  (V transposed for PV GEMM)
// ---------------------------------------------------------------------------
template <int MODE>
__global__ __launch_bounds__(256) void gemm_bf16(
    const __bf16* __restrict__ X, const __bf16* __restrict__ W,
    void* __restrict__ out) {
  constexpr int N = DD, K = DD;
  const int lane = threadIdx.x & 31;
  // wave id is uniform: force it scalar so bases live in SGPRs.
  const int wid  = __builtin_amdgcn_readfirstlane(
      blockIdx.x * 8 + (threadIdx.x >> 5));
  constexpr int WN = N / 64;                 // 16 waves across N
  const int wm = wid / WN;                   // 0..127
  const int wn = wid % WN;
  const int m0 = wm * 64, n0 = wn * 64;
  const int half = lane >> 4, lr = lane & 15;

  v8f acc[4][4] = {};

  const __bf16* arow[4];
  const __bf16* brow[4];
#pragma unroll
  for (int mt = 0; mt < 4; ++mt)
    arow[mt] = X + (size_t)(m0 + 16 * mt + lr) * K;
#pragma unroll
  for (int nt = 0; nt < 4; ++nt)
    brow[nt] = W + (size_t)(n0 + 16 * nt + lr) * K;

  for (int kk = 0; kk < K; kk += 32) {
    // A fragments (16x32, ISA 16-bit A layout): two contiguous 16B chunks.
    V16 a[4];
#pragma unroll
    for (int mt = 0; mt < 4; ++mt) {
      a[mt].h[0] = *(const v8bf*)(arow[mt] + kk + 8 * half);
      a[mt].h[1] = *(const v8bf*)(arow[mt] + kk + 16 + 8 * half);
      if (kk + 256 < K) __builtin_prefetch(arow[mt] + kk + 256, 0, 3);
    }
    // B fragments (32x16): lane = column n; W row n is contiguous in K.
#pragma unroll
    for (int nt = 0; nt < 4; ++nt) {
      V16 b;
      const __bf16* p = brow[nt] + kk + 8 * half;
      b.h[0] = *(const v8bf*)p;
      b.h[1] = *(const v8bf*)(p + 16);
      if (kk + 256 < K) __builtin_prefetch(brow[nt] + kk + 256, 0, 3);
#pragma unroll
      for (int mt = 0; mt < 4; ++mt)
        acc[mt][nt] = __builtin_amdgcn_wmma_f32_16x16x32_bf16(
            false, a[mt].v, false, b.v, (short)0, acc[mt][nt], false, false);
    }
  }

  // Store: D layout -> element i of lane = (m = 16*mt + i + 8*half, n = 16*nt + lr)
#pragma unroll
  for (int mt = 0; mt < 4; ++mt) {
#pragma unroll
    for (int nt = 0; nt < 4; ++nt) {
#pragma unroll
      for (int i = 0; i < 8; ++i) {
        const int m = m0 + 16 * mt + i + 8 * half;
        const int n = n0 + 16 * nt + lr;
        const float val = acc[mt][nt][i];
        if (MODE == 0) {
          ((float*)out)[(size_t)m * N + n] = val;
        } else {
          const int bb = m / SS, s = m % SS;
          const int hh = n / HD, hd = n % HD;
          unsigned short* o = (unsigned short*)out;
          if (MODE == 1)
            o[((size_t)(bb * HH + hh) * SS + s) * HD + hd] = f2bf(val);
          else
            o[((size_t)(bb * HH + hh) * HD + hd) * SS + s] = f2bf(val);
        }
      }
    }
  }
}

// ---------------------------------------------------------------------------
// Flash attention: one wave handles one (b, h, 16-query tile).
//   Streams KV in chunks of 32. All 4 K fragments and all 4 V fragments are
//   loaded up-front each chunk into distinct registers: score WMMAs wait only
//   for the K loads (in-order loadcnt), while the V loads ride out the whole
//   softmax VALU/trans section for free. Online softmax kept in the C/D
//   register layout, base-2 domain (hardware v_exp_f32 is exp2); row
//   reductions via shfl_xor stay inside the 16-lane halves that own rows
//   0-7 / 8-15. P re-swizzled D-layout -> A-layout through a 1KB per-wave
//   LDS buffer (single-wave DS ops are in-order; no barrier needed).
//   PV: 4 WMMA against pre-transposed V [B,H,HD,S].
// ---------------------------------------------------------------------------
__global__ __launch_bounds__(256) void attn_fwd(
    const __bf16* __restrict__ q, const __bf16* __restrict__ k,
    const __bf16* __restrict__ vt, unsigned short* __restrict__ ctx) {
  __shared__ unsigned short plds[8][16 * 32];

  const int lane = threadIdx.x & 31;
  const int wave = threadIdx.x >> 5;
  // (b, h, qt) are wave-uniform: force scalar for SALU control flow and
  // SGPR base addresses.
  const int wid = __builtin_amdgcn_readfirstlane(blockIdx.x * 8 + wave);
  const int qt = wid & 127;
  const int h  = (wid >> 7) & 15;
  const int b  = wid >> 11;
  const int half = lane >> 4, lr = lane & 15;

  const size_t bh = (size_t)(b * HH + h);
  const __bf16* qp = q + (bh * SS + (size_t)qt * 16) * HD;
  const __bf16* kp = k + bh * SS * HD;
  const __bf16* vp = vt + bh * HD * SS;
  unsigned short* pl = plds[wave];

  // Q A-fragments (16 rows x 64 K -> two 16x32 fragments), loaded once.
  V16 aq[2];
#pragma unroll
  for (int t = 0; t < 2; ++t) {
    const __bf16* r = qp + (size_t)lr * HD + 32 * t + 8 * half;
    aq[t].h[0] = *(const v8bf*)r;
    aq[t].h[1] = *(const v8bf*)(r + 16);
  }

  v8f o[4] = {};
  float mrow[8], lrow[8];
#pragma unroll
  for (int i = 0; i < 8; ++i) { mrow[i] = -3.0e38f; lrow[i] = 0.f; }
  // 1/sqrt(HD) * log2(e): softmax carried in base-2 domain.
  const float scale2 = 0.125f * 1.44269504088896340736f;
  const int kvend = (qt + 1) * 16; // causal bound (exclusive), uniform

  for (int kv0 = 0; kv0 < kvend; kv0 += 32) {
    // ---- K B-fragments (4 x 16B-pair), loaded first: WMMAs wait on these ----
    V16 bk[2][2];
#pragma unroll
    for (int t = 0; t < 2; ++t) {
      const __bf16* col = kp + (size_t)(kv0 + 16 * t + lr) * HD;
#pragma unroll
      for (int kkk = 0; kkk < 2; ++kkk) {
        const __bf16* p = col + 32 * kkk + 8 * half;
        bk[t][kkk].h[0] = *(const v8bf*)p;
        bk[t][kkk].h[1] = *(const v8bf*)(p + 16);
      }
    }
    // ---- V B-fragments issued now, consumed only after softmax ----
    V16 bv[4];
#pragma unroll
    for (int td = 0; td < 4; ++td) {
      const __bf16* p = vp + (size_t)(16 * td + lr) * SS + kv0 + 8 * half;
      bv[td].h[0] = *(const v8bf*)p;
      bv[td].h[1] = *(const v8bf*)(p + 16);
    }
    // Prefetch next chunk (gfx1250 global_prefetch); uniform branch.
    if (kv0 + 32 < kvend) {
      __builtin_prefetch(kp + (size_t)(kv0 + 32 + lane) * HD, 0, 3);
      __builtin_prefetch(vp + (size_t)lane * SS + kv0 + 32, 0, 3);
      __builtin_prefetch(vp + (size_t)(lane + 32) * SS + kv0 + 32, 0, 3);
    }

    // ---- scores: S = Q @ K^T, two n-tiles of 16 kv columns each ----
    v8f s[2] = {};
#pragma unroll
    for (int t = 0; t < 2; ++t)
#pragma unroll
      for (int kkk = 0; kkk < 2; ++kkk)
        s[t] = __builtin_amdgcn_wmma_f32_16x16x32_bf16(
            false, aq[kkk].v, false, bk[t][kkk].v, (short)0, s[t], false, false);

    // ---- scale (into base-2) + causal mask ----
    const int qbase = qt * 16 + 8 * half;   // element i -> query row qbase + i
#pragma unroll
    for (int t = 0; t < 2; ++t) {
      const int kv = kv0 + 16 * t + lr;
#pragma unroll
      for (int i = 0; i < 8; ++i) {
        const float v = s[t][i] * scale2;
        s[t][i] = (kv <= qbase + i) ? v : -3.0e38f;
      }
    }

    // ---- online softmax (base-2) ----
    float cm[8];
#pragma unroll
    for (int i = 0; i < 8; ++i) cm[i] = fmaxf(s[0][i], s[1][i]);
#pragma unroll
    for (int off = 1; off < 16; off <<= 1)
#pragma unroll
      for (int i = 0; i < 8; ++i)
        cm[i] = fmaxf(cm[i], __shfl_xor(cm[i], off, 32));

    float alpha[8];
#pragma unroll
    for (int i = 0; i < 8; ++i) {
      const float mn = fmaxf(mrow[i], cm[i]);
      alpha[i] = exp2f(mrow[i] - mn);
      mrow[i] = mn;
    }

    float rs[8];
#pragma unroll
    for (int i = 0; i < 8; ++i) {
      const float p0 = exp2f(s[0][i] - mrow[i]);
      const float p1 = exp2f(s[1][i] - mrow[i]);
      s[0][i] = p0; s[1][i] = p1;
      rs[i] = p0 + p1;
    }
#pragma unroll
    for (int off = 1; off < 16; off <<= 1)
#pragma unroll
      for (int i = 0; i < 8; ++i) rs[i] += __shfl_xor(rs[i], off, 32);
#pragma unroll
    for (int i = 0; i < 8; ++i) lrow[i] = lrow[i] * alpha[i] + rs[i];
#pragma unroll
    for (int td = 0; td < 4; ++td)
#pragma unroll
      for (int i = 0; i < 8; ++i) o[td][i] *= alpha[i];

    // ---- P: D-layout -> LDS row-major [16 q][32 kv] -> A-layout frag ----
#pragma unroll
    for (int t = 0; t < 2; ++t)
#pragma unroll
      for (int i = 0; i < 8; ++i)
        pl[(i + 8 * half) * 32 + 16 * t + lr] = f2bf(s[t][i]);

    V16 ap;
    const __bf16* pb = (const __bf16*)pl + (size_t)lr * 32;
    ap.h[0] = *(const v8bf*)(pb + 8 * half);
    ap.h[1] = *(const v8bf*)(pb + 16 + 8 * half);

    // ---- O += P @ V; V fragments have been in flight since loop top ----
#pragma unroll
    for (int td = 0; td < 4; ++td)
      o[td] = __builtin_amdgcn_wmma_f32_16x16x32_bf16(
          false, ap.v, false, bv[td].v, (short)0, o[td], false, false);
  }

  // ---- epilogue: ctx[b, s, h*HD + hd] = O / l  (bf16, row-major [B,S,D]) ----
#pragma unroll
  for (int td = 0; td < 4; ++td) {
    const int hd = 16 * td + lr;
#pragma unroll
    for (int i = 0; i < 8; ++i) {
      const int srow = qt * 16 + i + 8 * half;
      const float val = o[td][i] / lrow[i];
      ctx[((size_t)b * SS + srow) * DD + h * HD + hd] = f2bf(val);
    }
  }
}

// ---------------------------------------------------------------------------
// Launcher
// ---------------------------------------------------------------------------
extern "C" void kernel_launch(void* const* d_in, const int* in_sizes, int n_in,
                              void* d_out, int out_size, void* d_ws, size_t ws_size,
                              hipStream_t stream) {
  const float* x  = (const float*)d_in[0];
  const float* Wq = (const float*)d_in[1];
  const float* Wk = (const float*)d_in[2];
  const float* Wv = (const float*)d_in[3];
  const float* Wo = (const float*)d_in[4];

  char* ws = (char*)d_ws;
  const size_t xbytes = (size_t)MM * DD * 2;   // 16 MB
  const size_t wbytes = (size_t)DD * DD * 2;   //  2 MB
  unsigned short* xb  = (unsigned short*)(ws);
  unsigned short* wqb = (unsigned short*)(ws + xbytes);
  unsigned short* wkb = (unsigned short*)(ws + xbytes + wbytes);
  unsigned short* wvb = (unsigned short*)(ws + xbytes + 2 * wbytes);
  unsigned short* wob = (unsigned short*)(ws + xbytes + 3 * wbytes);
  unsigned short* qb  = (unsigned short*)(ws + xbytes + 4 * wbytes);
  unsigned short* kb  = (unsigned short*)(ws + 2 * xbytes + 4 * wbytes);
  unsigned short* vtb = (unsigned short*)(ws + 3 * xbytes + 4 * wbytes);
  unsigned short* ctxb= (unsigned short*)(ws + 4 * xbytes + 4 * wbytes);

  const int nx = MM * DD;        // 8388608
  const int nw = DD * DD;        // 1048576
  cast_f32_to_bf16<<<nx / 256, 256, 0, stream>>>(x, xb, nx);
  cast_f32_to_bf16<<<nw / 256, 256, 0, stream>>>(Wq, wqb, nw);
  cast_f32_to_bf16<<<nw / 256, 256, 0, stream>>>(Wk, wkb, nw);
  cast_f32_to_bf16<<<nw / 256, 256, 0, stream>>>(Wv, wvb, nw);
  cast_f32_to_bf16<<<nw / 256, 256, 0, stream>>>(Wo, wob, nw);

  // 64x64 tiles/wave -> (8192/64)*(1024/64) = 2048 waves -> 256 blocks of 8.
  gemm_bf16<1><<<256, 256, 0, stream>>>((const __bf16*)xb, (const __bf16*)wqb, qb);
  gemm_bf16<1><<<256, 256, 0, stream>>>((const __bf16*)xb, (const __bf16*)wkb, kb);
  gemm_bf16<2><<<256, 256, 0, stream>>>((const __bf16*)xb, (const __bf16*)wvb, vtb);

  // 8192 (b,h,qtile) waves -> 1024 blocks of 8 waves.
  attn_fwd<<<1024, 256, 0, stream>>>((const __bf16*)qb, (const __bf16*)kb,
                                     (const __bf16*)vtb, ctxb);

  gemm_bf16<0><<<256, 256, 0, stream>>>((const __bf16*)ctxb, (const __bf16*)wob,
                                        d_out);
}
